// FullyConnected_60060822667472
// MI455X (gfx1250) — compile-verified, matrix operand order checked by reading, not run on hardware
//
#include <hip/hip_runtime.h>
#include <stdint.h>

// c[h,n,m] = LeakyReLU( sum_d z0[n,d]*W[h,d]*z1[m,d] + bias[h], 0.2 )
// Per block: one 128x128 (n,m) tile x 16 h-values.
//   Ws (LDS, f32) : W h-tile (16x256), loaded once via TDM tensor_load_to_lds.
//   Bs (LDS, bf16): z1 tile, staged once (f32->bf16 convert).
//   As (LDS, bf16): z0 tile scaled by W[h,:], restaged per h (z0 is L2-resident).
//   v_wmma_f32_16x16x32_bf16, f32 accumulate, K = D = 256 (8 k-steps).

typedef __attribute__((ext_vector_type(16))) __bf16   v16bf;
typedef __attribute__((ext_vector_type(8)))  __bf16   v8bf;
typedef __attribute__((ext_vector_type(8)))  float    v8f;
typedef __attribute__((ext_vector_type(4)))  uint32_t su4;
typedef __attribute__((ext_vector_type(8)))  uint32_t su8;

constexpr int Nn = 1024, Mm = 1024, Dd = 256, Hh = 128;
constexpr int BN = 128, BM = 128, HT = 16;
constexpr int LDT = 264;                          // bf16 LDS row stride (528 B, 16B-aligned)
constexpr int LDW = 256;                          // f32 W row stride (TDM writes unpadded)
constexpr int BS_OFF = 0;
constexpr int AS_OFF = BN * LDT * 2;              // 67584
constexpr int WS_OFF = AS_OFF + BN * LDT * 2;     // 135168
constexpr int SMEM_BYTES = WS_OFF + HT * LDW * 4; // 151552 (<320KB, 2 WGs/WGP)
constexpr float SLOPE = 0.2f;

__device__ __forceinline__ unsigned f2bf(float f) {          // f32 -> bf16 (RNE)
  unsigned u = __float_as_uint(f);
  u += 0x7FFFu + ((u >> 16) & 1u);
  return u >> 16;
}
__device__ __forceinline__ unsigned pk2(float a, float b) {  // pack 2x bf16
  return f2bf(a) | (f2bf(b) << 16);
}

__global__ __launch_bounds__(256) void einsum_leaky_kernel(
    const float* __restrict__ z0, const float* __restrict__ z1,
    const float* __restrict__ W,  const float* __restrict__ bias,
    float* __restrict__ out)
{
  extern __shared__ __align__(16) char smem[];
  __bf16* Bs = (__bf16*)(smem + BS_OFF);
  __bf16* As = (__bf16*)(smem + AS_OFF);
  float*  Ws = (float*) (smem + WS_OFF);

  const int tid  = threadIdx.x;
  const int lane = tid & 31;
  const int wv   = tid >> 5;        // wave 0..7
  const int wr   = wv & 3;          // 32-row group within 128
  const int wc   = wv >> 2;         // 64-col group within 128
  const int lrow = lane & 15;       // fragment row/col within 16
  const int koff = (lane >> 4) * 8; // K sub-chunk per lane half

  const int m0 = blockIdx.x * BM;
  const int n0 = blockIdx.y * BN;
  const int h0 = blockIdx.z * HT;

  // ---- TDM: DMA the W h-tile (16x256 f32, contiguous = 4096-elem 1-D tile)
  //      into LDS at WS_OFF. One wave issues; barrier below publishes it.
  if (wv == 0) {
    const uint64_t ga = (uint64_t)(uintptr_t)(W + (size_t)h0 * Dd);
    const uint32_t elems = (uint32_t)(HT * Dd);   // 4096
    su4 g0;
    g0[0] = 1u;                                          // count=1, user mode
    g0[1] = (uint32_t)WS_OFF;                            // lds_addr (bytes)
    g0[2] = (uint32_t)ga;                                // global_addr[31:0]
    g0[3] = ((uint32_t)(ga >> 32) & 0x01FFFFFFu)         // global_addr[56:32]
            | (2u << 30);                                // type = 2 ("image")
    su8 g1;
    g1[0] = 2u << 16;                                    // data_size = 4 bytes
    g1[1] = (elems & 0xFFFFu) << 16;                     // tensor_dim0[15:0]
    g1[2] = (elems >> 16) | (1u << 16);                  // tensor_dim0[31:16] | tensor_dim1=1
    g1[3] = (elems & 0xFFFFu) << 16;                     // tile_dim0 = 4096
    g1[4] = 0u;                                          // tile_dim1/2 unused
    g1[5] = elems;                                       // tensor_dim0_stride
    g1[6] = 0u;
    g1[7] = 0u;
    asm volatile("tensor_load_to_lds %0, %1" :: "s"(g0), "s"(g1) : "memory");
    __builtin_amdgcn_s_wait_tensorcnt(0);
  }

  // ---- stage z1 tile -> Bs (bf16); Bs[m_local][k]
  {
    const int r  = tid >> 1;
    const int cb = (tid & 1) * 128;
    const float* src = z1 + (size_t)(m0 + r) * Dd + cb;
    __bf16* dst = Bs + r * LDT + cb;
    #pragma unroll
    for (int c = 0; c < 128; c += 4) {
      const float4 v = *(const float4*)(src + c);
      uint2 p; p.x = pk2(v.x, v.y); p.y = pk2(v.z, v.w);
      *(uint2*)(dst + c) = p;
    }
  }

  const int rA  = tid >> 1;
  const int cbA = (tid & 1) * 128;
  const float* z0src = z0 + (size_t)(n0 + rA) * Dd + cbA;

  for (int hl = 0; hl < HT; ++hl) {
    __syncthreads();  // Ws (TDM) / Bs visible (iter 0); prior As reads done

    // ---- stage W[h]-scaled z0 tile -> As (bf16); As[n_local][k]
    {
      const float* wsrow = Ws + hl * LDW + cbA;   // broadcast reads (2 addrs per wave)
      __bf16* dst = As + rA * LDT + cbA;
      #pragma unroll
      for (int c = 0; c < 128; c += 4) {
        const float4 v = *(const float4*)(z0src + c);
        const float4 w = *(const float4*)(wsrow + c);
        uint2 p;
        p.x = pk2(v.x * w.x, v.y * w.y);
        p.y = pk2(v.z * w.z, v.w * w.w);
        *(uint2*)(dst + c) = p;
      }
    }
    __syncthreads();

    // ---- 32x64 per-wave tile: 2x4 accumulators, K-loop over D
    v8f acc[2][4] = {};
    for (int kb = 0; kb < Dd; kb += 32) {
      v16bf a[2], b[4];
      #pragma unroll
      for (int ti = 0; ti < 2; ++ti) {
        const __bf16* p = As + (wr * 32 + ti * 16 + lrow) * LDT + kb + koff;
        const v8bf lo = *(const v8bf*)p;
        const v8bf hi = *(const v8bf*)(p + 16);
        a[ti] = __builtin_shufflevector(lo, hi, 0,1,2,3,4,5,6,7,
                                               8,9,10,11,12,13,14,15);
      }
      #pragma unroll
      for (int tj = 0; tj < 4; ++tj) {
        const __bf16* p = Bs + (wc * 64 + tj * 16 + lrow) * LDT + kb + koff;
        const v8bf lo = *(const v8bf*)p;
        const v8bf hi = *(const v8bf*)(p + 16);
        b[tj] = __builtin_shufflevector(lo, hi, 0,1,2,3,4,5,6,7,
                                               8,9,10,11,12,13,14,15);
      }
      #pragma unroll
      for (int ti = 0; ti < 2; ++ti)
        #pragma unroll
        for (int tj = 0; tj < 4; ++tj)
          acc[ti][tj] = __builtin_amdgcn_wmma_f32_16x16x32_bf16(
              false, a[ti], false, b[tj], (short)0, acc[ti][tj], false, false);
    }

    // ---- epilogue: bias + LeakyReLU + store
    const int   h  = h0 + hl;
    const float bv = bias[h];
    #pragma unroll
    for (int ti = 0; ti < 2; ++ti) {
      #pragma unroll
      for (int tj = 0; tj < 4; ++tj) {
        const int mcol  = m0 + wc * 64 + tj * 16 + lrow;
        const int nbase = n0 + wr * 32 + ti * 16 + ((lane >> 4) << 3);
        float* op = out + ((size_t)h * Nn + nbase) * (size_t)Mm + mcol;
        #pragma unroll
        for (int r = 0; r < 8; ++r) {
          const float v = acc[ti][tj][r] + bv;
          op[(size_t)r * Mm] = (v >= 0.0f) ? v : SLOPE * v;
        }
      }
    }
  }
}

extern "C" void kernel_launch(void* const* d_in, const int* in_sizes, int n_in,
                              void* d_out, int out_size, void* d_ws, size_t ws_size,
                              hipStream_t stream) {
  (void)in_sizes; (void)n_in; (void)d_ws; (void)ws_size; (void)out_size;
  const float* z0 = (const float*)d_in[0];
  const float* z1 = (const float*)d_in[1];
  const float* W  = (const float*)d_in[2];
  const float* bs = (const float*)d_in[3];
  float* out = (float*)d_out;

  static_assert(SMEM_BYTES <= 320 * 1024, "LDS budget");
  (void)hipFuncSetAttribute(reinterpret_cast<const void*>(einsum_leaky_kernel),
                            hipFuncAttributeMaxDynamicSharedMemorySize,
                            SMEM_BYTES);
  dim3 grid(Mm / BM, Nn / BN, Hh / HT);  // 8 x 8 x 8 = 512 blocks
  einsum_leaky_kernel<<<grid, dim3(256), SMEM_BYTES, stream>>>(z0, z1, W, bs, out);
}
